// ParallelHybridReasoningLayer_34342558498999
// MI455X (gfx1250) — matrix operand order repaired and checked by reading, not compiled, not run on hardware
//
#include <hip/hip_runtime.h>
#include <hip/hip_bf16.h>

typedef __attribute__((ext_vector_type(16))) _Float16 v16h;
typedef __attribute__((ext_vector_type(8)))  float    v8f;

#define WMMA_F16(A, B, C) \
  __builtin_amdgcn_wmma_f32_16x16x32_f16(false, (A), false, (B), (short)0, (C), false, false)

__device__ __forceinline__ float gelu_exact(float v) {
  return 0.5f * v * (1.0f + erff(v * 0.70710678118654752f));
}

// Pack 16 contiguous floats (four float4) into the f16 B-operand layout slot.
__device__ __forceinline__ v16h load_b16(const float* __restrict__ p) {
  const float4* q = (const float4*)p;
  float4 b0 = q[0], b1 = q[1], b2 = q[2], b3 = q[3];
  v16h b;
  b[0]=(_Float16)b0.x;  b[1]=(_Float16)b0.y;  b[2]=(_Float16)b0.z;  b[3]=(_Float16)b0.w;
  b[4]=(_Float16)b1.x;  b[5]=(_Float16)b1.y;  b[6]=(_Float16)b1.z;  b[7]=(_Float16)b1.w;
  b[8]=(_Float16)b2.x;  b[9]=(_Float16)b2.y;  b[10]=(_Float16)b2.z; b[11]=(_Float16)b2.w;
  b[12]=(_Float16)b3.x; b[13]=(_Float16)b3.y; b[14]=(_Float16)b3.z; b[15]=(_Float16)b3.w;
  return b;
}

// Pack two 8-float contiguous runs (p0 = K-lo run, p1 = K-hi run) into A-operand layout.
__device__ __forceinline__ v16h load_a16(const float* __restrict__ p0,
                                         const float* __restrict__ p1, float scale) {
  const float4* q0 = (const float4*)p0;
  const float4* q1 = (const float4*)p1;
  float4 a0 = q0[0], a1 = q0[1], a2 = q1[0], a3 = q1[1];
  v16h a;
  a[0]=(_Float16)(a0.x*scale);  a[1]=(_Float16)(a0.y*scale);
  a[2]=(_Float16)(a0.z*scale);  a[3]=(_Float16)(a0.w*scale);
  a[4]=(_Float16)(a1.x*scale);  a[5]=(_Float16)(a1.y*scale);
  a[6]=(_Float16)(a1.z*scale);  a[7]=(_Float16)(a1.w*scale);
  a[8]=(_Float16)(a2.x*scale);  a[9]=(_Float16)(a2.y*scale);
  a[10]=(_Float16)(a2.z*scale); a[11]=(_Float16)(a2.w*scale);
  a[12]=(_Float16)(a3.x*scale); a[13]=(_Float16)(a3.y*scale);
  a[14]=(_Float16)(a3.z*scale); a[15]=(_Float16)(a3.w*scale);
  return a;
}

// ---------------------------------------------------------------------------
// WMMA GEMM:  C[M,N] = epi( A[M,K] @ W[N,K]^T + bias[N] (+ resid) )
// One wave computes a 16x64 strip (4 N-tiles share one A operand).
// Block = 256 threads = 8 waves.
// ---------------------------------------------------------------------------
__global__ void __launch_bounds__(256) gemm_wmma_f16(
    const float* __restrict__ A, const float* __restrict__ W,
    const float* __restrict__ bias, const float* __restrict__ resid,
    float* __restrict__ C, int M, int N, int K, int applyGelu)
{
  const int lane   = threadIdx.x & 31;
  const int wave   = threadIdx.x >> 5;
  const int l16    = lane & 15;
  const int half16 = (lane >= 16) ? 1 : 0;

  const int nT   = N >> 6;                 // 64-wide strips
  const int tile = blockIdx.x * 8 + wave;
  const int tm   = (tile / nT) << 4;
  const int tn   = (tile % nT) << 6;

  const float* __restrict__ Arow = A + (size_t)(tm + l16) * K;
  const float* __restrict__ Wr0  = W + (size_t)(tn      + l16) * K;
  const float* __restrict__ Wr1  = W + (size_t)(tn + 16 + l16) * K;
  const float* __restrict__ Wr2  = W + (size_t)(tn + 32 + l16) * K;
  const float* __restrict__ Wr3  = W + (size_t)(tn + 48 + l16) * K;

  v8f acc[4]; acc[0] = {}; acc[1] = {}; acc[2] = {}; acc[3] = {};

  for (int kb = 0; kb < K; kb += 32) {
    // A operand: K-lo run at kb + 16*half16, K-hi run at kb + 16*half16 + 8
    const v16h a = load_a16(Arow + kb + (half16 << 4),
                            Arow + kb + 8 + (half16 << 4), 1.0f);
    const int wo = kb + (half16 << 4);     // B operand: 16 contiguous
    acc[0] = WMMA_F16(a, load_b16(Wr0 + wo), acc[0]);
    acc[1] = WMMA_F16(a, load_b16(Wr1 + wo), acc[1]);
    acc[2] = WMMA_F16(a, load_b16(Wr2 + wo), acc[2]);
    acc[3] = WMMA_F16(a, load_b16(Wr3 + wo), acc[3]);
  }

#pragma unroll
  for (int nt = 0; nt < 4; ++nt) {
#pragma unroll
    for (int r = 0; r < 8; ++r) {
      const int m = tm + r + (half16 << 3);
      const int n = tn + (nt << 4) + l16;
      float v = acc[nt][r] + bias[n];
      if (applyGelu) v = gelu_exact(v);
      if (resid) v += resid[(size_t)m * N + n];
      C[(size_t)m * N + n] = v;
    }
  }
}

// ---------------------------------------------------------------------------
// Causal flash attention. qkv rows = [q(256) | k(256) | v(256)], H=4, hd=64.
// One wave (block=32) handles 16 query rows of one (b,h); keys in chunks of 32.
// ---------------------------------------------------------------------------
__global__ void __launch_bounds__(32) attn_wmma(
    const float* __restrict__ qkv, float* __restrict__ attnO)
{
  __shared__ float pbuf[16][32];

  const int lane   = threadIdx.x & 31;
  const int l16    = lane & 15;
  const int half16 = (lane >= 16) ? 1 : 0;

  const int id = blockIdx.x;        // 0..2047
  const int bh = id >> 6;           // (b*H + h)
  const int qt = (id & 63) << 4;    // query tile start
  const int b  = bh >> 2;
  const int hh = bh & 3;

  const float* __restrict__ base = qkv + (size_t)b * 1024 * 768;

  // Q pre-scaled by 1/sqrt(hd)=1/8 as two A operands (hd-dims 0..31, 32..63)
  const float* Qrow = base + (size_t)(qt + l16) * 768 + hh * 64;
  const v16h aQ0 = load_a16(Qrow + (half16 << 4), Qrow + 8 + (half16 << 4), 0.125f);
  const v16h aQ1 = load_a16(Qrow + 32 + (half16 << 4), Qrow + 40 + (half16 << 4), 0.125f);

  v8f o[4]; o[0] = {}; o[1] = {}; o[2] = {}; o[3] = {};
  float mrow[8], lrow[8];
#pragma unroll
  for (int r = 0; r < 8; ++r) { mrow[r] = -1e30f; lrow[r] = 0.0f; }

  const int nChunks = (qt + 16 + 31) >> 5;
  for (int c = 0; c < nChunks; ++c) {
    const int ks = c << 5;

    // S = Q @ K^T  (two 16x16 key tiles, K-dim = hd = 64)
    v8f s0 = {}, s1 = {};
    {
      const float* K0 = base + (size_t)(ks + l16) * 768 + 256 + hh * 64;
      const float* K1 = base + (size_t)(ks + 16 + l16) * 768 + 256 + hh * 64;
      const int ko = (half16 << 4);
      s0 = WMMA_F16(aQ0, load_b16(K0 + ko),      s0);
      s0 = WMMA_F16(aQ1, load_b16(K0 + 32 + ko), s0);
      s1 = WMMA_F16(aQ0, load_b16(K1 + ko),      s1);
      s1 = WMMA_F16(aQ1, load_b16(K1 + 32 + ko), s1);
    }

    // Causal mask + online softmax (row = 16-lane group reduction)
#pragma unroll
    for (int r = 0; r < 8; ++r) {
      const int qrow = qt + r + (half16 << 3);
      if (ks + l16 > qrow)      s0[r] = -1e30f;
      if (ks + 16 + l16 > qrow) s1[r] = -1e30f;

      float mx = fmaxf(s0[r], s1[r]);
#pragma unroll
      for (int sm = 8; sm >= 1; sm >>= 1) mx = fmaxf(mx, __shfl_xor(mx, sm, 32));
      const float mnew = fmaxf(mrow[r], mx);
      const float sc   = expf(mrow[r] - mnew);
      const float p0   = expf(s0[r] - mnew);
      const float p1   = expf(s1[r] - mnew);
      float ps = p0 + p1;
#pragma unroll
      for (int sm = 8; sm >= 1; sm >>= 1) ps += __shfl_xor(ps, sm, 32);
      lrow[r] = lrow[r] * sc + ps;
      mrow[r] = mnew;
      o[0][r] *= sc; o[1][r] *= sc; o[2][r] *= sc; o[3][r] *= sc;

      const int row = r + (half16 << 3);
      pbuf[row][l16]      = p0;
      pbuf[row][16 + l16] = p1;
    }
    __syncthreads();  // single-wave block: LDS write->read ordering

    // Re-gather P as f16 A operand (16 rows x 32 keys) via b128 LDS reads
    const v16h aP = load_a16(&pbuf[l16][(half16 << 4)],
                             &pbuf[l16][8 + (half16 << 4)], 1.0f);
    __syncthreads();  // protect pbuf before next chunk overwrites

    // O += P @ V  (four 16-col tiles of hd, K = 32 keys)
#pragma unroll
    for (int nt = 0; nt < 4; ++nt) {
      v16h bv;
#pragma unroll
      for (int h = 0; h < 16; ++h) {
        const int key = ks + h + (half16 << 4);
        bv[h] = (_Float16)base[(size_t)key * 768 + 512 + hh * 64 + (nt << 4) + l16];
      }
      o[nt] = WMMA_F16(aP, bv, o[nt]);
    }
  }

#pragma unroll
  for (int r = 0; r < 8; ++r) {
    const float inv = 1.0f / lrow[r];
    const int m = b * 1024 + qt + r + (half16 << 3);
    float* dst = attnO + (size_t)m * 256 + hh * 64;
    dst[l16]      = o[0][r] * inv;
    dst[16 + l16] = o[1][r] * inv;
    dst[32 + l16] = o[2][r] * inv;
    dst[48 + l16] = o[3][r] * inv;
  }
}

// ---------------------------------------------------------------------------
// LayerNorm over D=256: one wave per row, 8 elems/lane, shuffle reductions.
// ---------------------------------------------------------------------------
__global__ void __launch_bounds__(256) layernorm_k(
    const float* __restrict__ in, float* __restrict__ out,
    const float* __restrict__ s, const float* __restrict__ bi,
    int rows, long inStride, long outStride)
{
  const int lane = threadIdx.x & 31;
  const int wave = threadIdx.x >> 5;
  const int row  = blockIdx.x * 8 + wave;
  if (row >= rows) return;

  const float* x = in + (size_t)row * inStride;
  float v[8]; float sum = 0.0f, sq = 0.0f;
#pragma unroll
  for (int i = 0; i < 8; ++i) { v[i] = x[lane + 32 * i]; sum += v[i]; sq += v[i] * v[i]; }
#pragma unroll
  for (int sm = 16; sm >= 1; sm >>= 1) { sum += __shfl_xor(sum, sm, 32); sq += __shfl_xor(sq, sm, 32); }
  const float mu   = sum * (1.0f / 256.0f);
  const float var  = sq * (1.0f / 256.0f) - mu * mu;
  const float rstd = rsqrtf(var + 1e-5f);
  float* y = out + (size_t)row * outStride;
#pragma unroll
  for (int i = 0; i < 8; ++i) {
    const int n = lane + 32 * i;
    y[n] = (v[i] - mu) * rstd * s[n] + bi[n];
  }
}

// Depthwise causal conv, window 8: out[b,t,d] = sum_j x[b,t-7+j,d]*w[d,j] + b[d]
__global__ void conv_k(const float* __restrict__ x, const float* __restrict__ w,
                       const float* __restrict__ bi, float* __restrict__ out)
{
  const int idx = blockIdx.x * 256 + threadIdx.x;  // m*256 + d
  const int d = idx & 255;
  const int m = idx >> 8;
  const int t = m & 1023;
  float acc = bi[d];
#pragma unroll
  for (int j = 0; j < 8; ++j) {
    if (t - 7 + j >= 0) acc += x[(size_t)(m - 7 + j) * 256 + d] * w[d * 8 + j];
  }
  out[(size_t)idx] = acc;
}

// EMA scan along T per (b,d) channel; also emits final trajectory summary.
__global__ void ema_k(const float* __restrict__ x, float* __restrict__ ema,
                      float* __restrict__ traj)
{
  const int idx = blockIdx.x * 256 + threadIdx.x;  // b*256 + d, total 2048
  const int b = idx >> 8, d = idx & 255;
  const float* xp = x + (size_t)b * 1024 * 256 + d;
  float* ep = ema + (size_t)b * 1024 * 256 + d;
  float c = 0.0f;
  for (int t = 0; t < 1024; ++t) {
    c = 0.9f * c + (1.0f - 0.9f) * xp[(size_t)t * 256];
    ep[(size_t)t * 256] = c;
  }
  traj[idx] = c;
}

// states = seq_state + cumsum(deltas)*0.5 per (b,d) channel.
__global__ void scan_k(const float* __restrict__ deltas, const float* __restrict__ s0,
                       float* __restrict__ states)
{
  const int idx = blockIdx.x * 256 + threadIdx.x;  // b*256 + d
  const int b = idx >> 8, d = idx & 255;
  const float* dp = deltas + (size_t)b * 1024 * 256 + d;
  float* sp = states + (size_t)b * 1024 * 256 + d;
  const float base = s0[idx];
  float acc = 0.0f;
  for (int t = 0; t < 1024; ++t) {
    acc += dp[(size_t)t * 256];
    sp[(size_t)t * 256] = base + 0.5f * acc;
  }
}

// combined[m, :] = [x + 0.3*pg | local | ema]   (8192 x 768)
__global__ void combined_k(const float* __restrict__ x1, const float* __restrict__ pg,
                           const float* __restrict__ loc, const float* __restrict__ ema,
                           float* __restrict__ comb)
{
  const size_t idx = (size_t)blockIdx.x * 256 + threadIdx.x;
  const int col = (int)(idx % 768);
  const size_t m = idx / 768;
  float v;
  if (col < 256)      v = x1[m * 256 + col] + 0.3f * pg[m * 256 + col];
  else if (col < 512) v = loc[m * 256 + (col - 256)];
  else                v = ema[m * 256 + (col - 512)];
  comb[idx] = v;
}

__global__ void addscaled_k(const float* __restrict__ a, const float* __restrict__ bsrc,
                            float sc, float* __restrict__ out)
{
  const size_t i = (size_t)blockIdx.x * 256 + threadIdx.x;
  out[i] = a[i] + sc * bsrc[i];
}

// ---------------------------------------------------------------------------
extern "C" void kernel_launch(void* const* d_in, const int* in_sizes, int n_in,
                              void* d_out, int out_size, void* d_ws, size_t ws_size,
                              hipStream_t stream) {
  (void)in_sizes; (void)n_in; (void)out_size; (void)ws_size;

  const float* X        = (const float*)d_in[0];   // parallel_repr [8,1024,256]
  const float* SS       = (const float*)d_in[1];   // sequential_state [8,256]
  const float* in_w     = (const float*)d_in[3];
  const float* in_b     = (const float*)d_in[4];
  const float* op_w     = (const float*)d_in[5];
  const float* op_b     = (const float*)d_in[6];
  const float* conv_w   = (const float*)d_in[7];
  const float* conv_b   = (const float*)d_in[8];
  const float* p2s_w    = (const float*)d_in[9];
  const float* p2s_b    = (const float*)d_in[10];
  const float* s2p_w    = (const float*)d_in[11];
  const float* s2p_b    = (const float*)d_in[12];
  const float* tw1      = (const float*)d_in[13];
  const float* tb1      = (const float*)d_in[14];
  const float* tw2      = (const float*)d_in[15];
  const float* tb2      = (const float*)d_in[16];
  const float* fw1      = (const float*)d_in[17];
  const float* fb1      = (const float*)d_in[18];
  const float* fw2      = (const float*)d_in[19];
  const float* fb2      = (const float*)d_in[20];
  const float* ln1_s    = (const float*)d_in[21];
  const float* ln1_b    = (const float*)d_in[22];
  const float* ln2_s    = (const float*)d_in[23];
  const float* ln2_b    = (const float*)d_in[24];
  const float* ln3_s    = (const float*)d_in[25];
  const float* ln3_b    = (const float*)d_in[26];
  const float* ln4_s    = (const float*)d_in[27];
  const float* ln4_b    = (const float*)d_in[28];

  float* out_x    = (float*)d_out;                 // [8,1024,256]
  float* out_seq  = out_x + 8 * 1024 * 256;        // [8,256]
  float* out_traj = out_seq + 8 * 256;             // [8,256]

  // Workspace layout (floats); regions reused across phases.
  float* ws   = (float*)d_ws;
  float* bufA = ws;                    // 8,388,608 : qkv -> combined -> f
  float* hbuf = bufA + 8388608;        // 4,194,304 : trans1 hidden
  float* x1   = hbuf + 4194304;        // 2,097,152 : post-LN1 x (persistent)
  float* b1   = x1 + 2097152;          // attn_out  -> seq_guidance
  float* b2   = b1 + 2097152;          // resid sum -> ffn_in -> ffn_out
  float* b4   = b2 + 2097152;          // local     -> states
  float* b5   = b4 + 2097152;          // ema (kept until combined)
  float* b6   = b5 + 2097152;          // parallel_guidance -> deltas

  const int M = 8192;
  // (M/16)*(N/64) strips, 8 waves (strips) per block of 256
  const int g768  = (M / 16) * (768 / 64) / 8;   // 768
  const int g512  = (M / 16) * (512 / 64) / 8;   // 512
  const int g256  = (M / 16) * (256 / 64) / 8;   // 256
  const int g1024 = (M / 16) * (1024 / 64) / 8;  // 1024

  // 1) qkv = X @ in_proj_w^T + b               [8192,768]
  gemm_wmma_f16<<<g768, 256, 0, stream>>>(X, in_w, in_b, nullptr, bufA, M, 768, 256, 0);
  // 2) causal attention                         [8192,256]
  attn_wmma<<<2048, 32, 0, stream>>>(bufA, b1);
  // 3) out_proj + residual (parallel_repr)
  gemm_wmma_f16<<<g256, 256, 0, stream>>>(b1, op_w, op_b, X, b2, M, 256, 256, 0);
  // 4) x = LN1
  layernorm_k<<<1024, 256, 0, stream>>>(b2, x1, ln1_s, ln1_b, M, 256, 256);
  // 5) depthwise conv -> local_history
  conv_k<<<8192, 256, 0, stream>>>(x1, conv_w, conv_b, b4);
  // 6) EMA -> global_summary (+ final trajectory output)
  ema_k<<<8, 256, 0, stream>>>(x1, b5, out_traj);
  // 7) parallel_guidance = x @ p2s^T + b
  gemm_wmma_f16<<<g256, 256, 0, stream>>>(x1, p2s_w, p2s_b, nullptr, b6, M, 256, 256, 0);
  // 8) combined = [x+0.3*pg | local | ema]      [8192,768]  (reuses qkv region)
  combined_k<<<24576, 256, 0, stream>>>(x1, b6, b4, b5, bufA);
  // 9) h = gelu(combined @ trans_w1^T + b1)     [8192,512]
  gemm_wmma_f16<<<g512, 256, 0, stream>>>(bufA, tw1, tb1, nullptr, hbuf, M, 512, 768, 1);
  // 10) deltas = h @ trans_w2^T + b2            [8192,256]
  gemm_wmma_f16<<<g256, 256, 0, stream>>>(hbuf, tw2, tb2, nullptr, b6, M, 256, 512, 0);
  // 11) states = seq_state + cumsum(deltas)*0.5
  scan_k<<<8, 256, 0, stream>>>(b6, SS, b4);
  // 12) states = LN3(states)  (in-place safe: per-row read-all-then-write)
  layernorm_k<<<1024, 256, 0, stream>>>(b4, b4, ln3_s, ln3_b, M, 256, 256);
  // 13) final_sequential_state = LN4(states[:, -1])
  layernorm_k<<<1, 256, 0, stream>>>(b4 + (size_t)1023 * 256, out_seq, ln4_s, ln4_b,
                                     8, (long)1024 * 256, 256);
  // 14) seq_guidance = states @ s2p^T + b
  gemm_wmma_f16<<<g256, 256, 0, stream>>>(b4, s2p_w, s2p_b, nullptr, b1, M, 256, 256, 0);
  // 15) ffn_in = x + 0.3*seq_guidance
  addscaled_k<<<8192, 256, 0, stream>>>(x1, b1, 0.3f, b2);
  // 16) f = gelu(ffn_in @ ffn_w1^T + b1)        [8192,1024] (reuses bufA region)
  gemm_wmma_f16<<<g1024, 256, 0, stream>>>(b2, fw1, fb1, nullptr, bufA, M, 1024, 256, 1);
  // 17) ffn_out = f @ ffn_w2^T + b2 + x (resid)
  gemm_wmma_f16<<<g256, 256, 0, stream>>>(bufA, fw2, fb2, x1, b2, M, 256, 1024, 0);
  // 18) x_out = LN2(...)
  layernorm_k<<<1024, 256, 0, stream>>>(b2, out_x, ln2_s, ln2_b, M, 256, 256);
}
